// ControlledAttentionBlock_63118839382520
// MI455X (gfx1250) — matrix operand order repaired and checked by reading, not compiled, not run on hardware
//
#include <hip/hip_runtime.h>
#include <hip/hip_fp16.h>

// ---------------------------------------------------------------------------
// ControlledAttentionBlock on MI455X (gfx1250, wave32, WMMA 16x16x32 f16)
//   B=16, L=512, D=1024, H=4, hd=256, DFF=4096
// GEMMs: 8-wave blocks, 128x64 C-tile, B staged in LDS via
// global_load_async_to_lds_b128 (ASYNCcnt double-buffered, peeled pipeline),
// A streamed + WGP-scope prefetch.
// ---------------------------------------------------------------------------

typedef __attribute__((ext_vector_type(16))) _Float16 v16h;
typedef __attribute__((ext_vector_type(8)))  float    v8f;
typedef __attribute__((ext_vector_type(4)))  int      i32x4;

#define AS1 __attribute__((address_space(1)))
#define AS3 __attribute__((address_space(3)))

#define B_   16
#define L_   512
#define D_   1024
#define H_   4
#define HD_  256
#define M_   (B_ * L_)     // 8192 rows
#define DFF_ 4096

#if __has_builtin(__builtin_amdgcn_global_load_async_to_lds_b128) && \
    __has_builtin(__builtin_amdgcn_s_wait_asynccnt)
#define ASYNC_LDS 1
#endif

static __device__ inline v8f wmma16(v16h a, v16h b, v8f c) {
  // v_wmma_f32_16x16x32_f16  (neg_a, A, neg_b, B, c_mod, C, reuse_a, reuse_b)
  return __builtin_amdgcn_wmma_f32_16x16x32_f16(false, a, false, b, (short)0, c,
                                                false, false);
}

// A fragment, 16x32 f16, row-major source with leading dim lda.
// ISA layout: lanes 0-15 row M=lane hold K {kb..kb+7, kb+16..kb+23} kb=0;
// lanes 16-31 same rows with kb=8.
static __device__ inline v16h load_a_frag(const _Float16* A, int lda) {
  const int lane = threadIdx.x & 31;
  const int r    = lane & 15;
  const int kb   = (lane >> 4) << 3;            // 0 or 8
  const _Float16* p = A + (size_t)r * lda + kb;
  v16h f;
#pragma unroll
  for (int i = 0; i < 8; ++i) { f[i] = p[i]; f[i + 8] = p[i + 16]; }
  return f;
}

// B fragment, 32x16 f16 (KxN): lane holds column N=lane%16, K {kb..kb+7, kb+16..kb+23}, kb=0 or 8.
static __device__ inline v16h load_b_frag_lds(const _Float16* Bl, int ldb,
                                              int n0) {
  const int lane = threadIdx.x & 31;
  const int kb   = (lane >> 4) << 3;
  const _Float16* p = Bl + (size_t)kb * ldb + n0 + (lane & 15);
  v16h f;
#pragma unroll
  for (int i = 0; i < 8; ++i) { f[i] = p[(size_t)i * ldb]; f[i + 8] = p[(size_t)(i + 16) * ldb]; }
  return f;
}

// ---------------------------------------------------------------------------
// Generic WMMA GEMM body: 256 threads = 8 waves; wave w owns rows
// (blockIdx.x*8+w)*16 .. +15; block owns cols blockIdx.y*64 .. +63.
// B tile [32k x 64n] staged once per block into LDS (double buffered),
// shared by all 8 waves. Steady-state loop is branch-free: stage(next),
// s_wait_asynccnt(1), barrier, 4x WMMA, barrier; last K-step peeled.
// ---------------------------------------------------------------------------
#define BPAD 8

template <bool BIAS, bool RELU, bool RES, bool O16, bool O32>
static __device__ inline void gemm_body(const _Float16* A, int lda,
                                        const _Float16* B, int ldb,
                                        const float* bias,
                                        const float* resid, int ldr,
                                        float* out32, int ldo,
                                        _Float16* out16, int ldo16, int K) {
  __shared__ _Float16 ldsB[2][32][64 + BPAD];

  const int wave = threadIdx.x >> 5;
  const int lane = threadIdx.x & 31;
  const int row0 = (blockIdx.x * 8 + wave) * 16;
  const int col0 = blockIdx.y * 64;

  // staging assignment: thread t moves 16B: k-row = t>>3, 8-half segment = t&7
  const int krow = threadIdx.x >> 3;
  const int seg  = (threadIdx.x & 7) * 8;

  auto stage = [&](int buf, int k0) {
    const _Float16* g = B + (size_t)(k0 + krow) * ldb + col0 + seg;
    _Float16* l = &ldsB[buf][krow][seg];
#ifdef ASYNC_LDS
    __builtin_amdgcn_global_load_async_to_lds_b128(
        (AS1 i32x4*)g, (AS3 i32x4*)l, 0, 0);
#else
    const int4 d = *(const int4*)g;
    *(int4*)l = d;
#endif
  };

  v8f acc[4] = {};
  const _Float16* Ap = A + (size_t)row0 * lda;

  stage(0, 0);
  int k0 = 0;
  // steady state: K-32 iterations, unconditional stage/wait
  for (; k0 + 32 < K; k0 += 32) {
    const int p = (k0 >> 5) & 1;
    stage(p ^ 1, k0 + 32);                  // fill next buffer
#ifdef ASYNC_LDS
    __builtin_amdgcn_s_wait_asynccnt(1);    // this buffer's fill retired
#endif
    __syncthreads();                        // all waves' fills visible
    __builtin_prefetch(Ap + k0 + 64, 0, 3); // next A K-slice, WGP scope
    v16h a = load_a_frag(Ap + k0, lda);
#pragma unroll
    for (int t = 0; t < 4; ++t) {
      v16h bf = load_b_frag_lds(&ldsB[p][0][0], 64 + BPAD, t * 16);
      acc[t] = wmma16(a, bf, acc[t]);
    }
    __syncthreads();                        // reads done before refill
  }
  // peeled last K-step
  {
#ifdef ASYNC_LDS
    __builtin_amdgcn_s_wait_asynccnt(0);
#endif
    __syncthreads();
    const int p = (k0 >> 5) & 1;
    v16h a = load_a_frag(Ap + k0, lda);
#pragma unroll
    for (int t = 0; t < 4; ++t) {
      v16h bf = load_b_frag_lds(&ldsB[p][0][0], 64 + BPAD, t * 16);
      acc[t] = wmma16(a, bf, acc[t]);
    }
  }

  const int colb  = lane & 15;
  const int rbase = (lane >> 4) << 3;   // 0 or 8
#pragma unroll
  for (int t = 0; t < 4; ++t) {
#pragma unroll
    for (int i = 0; i < 8; ++i) {
      const int r = row0 + rbase + i;
      const int c = col0 + t * 16 + colb;
      float v = acc[t][i];
      if (BIAS) v += bias[c];
      if (RELU) v = fmaxf(v, 0.0f);
      if (RES)  v += resid[(size_t)r * ldr + c];
      if (O32)  out32[(size_t)r * ldo  + c] = v;
      if (O16)  out16[(size_t)r * ldo16 + c] = (_Float16)v;
    }
  }
}

template <bool BIAS, bool RELU, bool RES, bool O16, bool O32>
__global__ __launch_bounds__(256) void k_gemm(const _Float16* A, int lda,
                                              const _Float16* B, int ldb,
                                              const float* bias,
                                              const float* resid, int ldr,
                                              float* out32, int ldo,
                                              _Float16* out16, int ldo16,
                                              int K) {
  gemm_body<BIAS, RELU, RES, O16, O32>(A, lda, B, ldb, bias, resid, ldr,
                                       out32, ldo, out16, ldo16, K);
}

// attn @ V, batched over (b,h) via blockIdx.z. probs16: [B*H, L, L] f16,
// V viewed per head as [L, hd] strided slice of the [M, D] f16 V buffer.
__global__ __launch_bounds__(256) void k_gemm_av(const _Float16* probs16,
                                                 const _Float16* v16,
                                                 _Float16* ao16) {
  const int z = blockIdx.z;
  const int b = z >> 2;
  const int h = z & 3;
  const _Float16* A  = probs16 + (size_t)z * L_ * L_;
  const _Float16* Bv = v16 + (size_t)b * L_ * D_ + h * HD_;
  _Float16*       O  = ao16 + (size_t)b * L_ * D_ + h * HD_;
  gemm_body<false, false, false, true, false>(A, L_, Bv, D_, nullptr, nullptr, 0,
                                              nullptr, 0, O, D_, L_);
}

// ---------------------------------------------------------------------------
// Fused scores + softmax. Block = 256 threads (8 waves) owns a 16 x 512
// strip of scores for one (b,h). K^T slice staged transposed through LDS.
// ---------------------------------------------------------------------------
__global__ __launch_bounds__(256) void k_attn(const _Float16* q16,
                                              const _Float16* k16,
                                              const float* cbias,
                                              const float* sharp,
                                              _Float16* probs16,
                                              float* attn0_out) {
  __shared__ _Float16 ldsB[32][L_ + 8];     // [k][n] transposed K slice
  __shared__ float    ldsS[16][L_ + 4];     // raw scores strip

  const int bh = blockIdx.y;
  const int b  = bh >> 2;
  const int h  = bh & 3;
  const int r0 = blockIdx.x * 16;
  const int wave = threadIdx.x >> 5;
  const int lane = threadIdx.x & 31;
  const int col0 = wave * 64;

  const _Float16* Q  = q16 + (size_t)b * L_ * D_ + (size_t)r0 * D_ + h * HD_;
  const _Float16* Kp = k16 + (size_t)b * L_ * D_ + h * HD_;

  v8f acc[4] = {};

  for (int k0 = 0; k0 < HD_; k0 += 32) {
    __syncthreads();
    // stage K[n][k0..k0+31] -> ldsB[k][n], 2 key rows per thread
#pragma unroll
    for (int rr = 0; rr < 2; ++rr) {
      const int n = threadIdx.x * 2 + rr;
      const _Float16* src = Kp + (size_t)n * D_ + k0;
#pragma unroll
      for (int j = 0; j < 32; ++j) ldsB[j][n] = src[j];
    }
    __syncthreads();

    v16h a = load_a_frag(Q + k0, D_);
#pragma unroll
    for (int t = 0; t < 4; ++t) {
      v16h bf = load_b_frag_lds(&ldsB[0][0], L_ + 8, col0 + t * 16);
      acc[t] = wmma16(a, bf, acc[t]);
    }
  }

  // scale (1/sqrt(256) = 1/16), row-0 collapse bias + sharpening, -> LDS
  const float sh   = sharp[0];
  const int  colb  = lane & 15;
  const int  rbase = (lane >> 4) << 3;
#pragma unroll
  for (int t = 0; t < 4; ++t) {
#pragma unroll
    for (int i = 0; i < 8; ++i) {
      const int row = rbase + i;
      const int col = col0 + t * 16 + colb;
      float v = acc[t][i] * 0.0625f;
      if (r0 + row == 0) v = (v + cbias[col]) * sh;
      ldsS[row][col] = v;
    }
  }
  __syncthreads();

  // softmax: each wave owns 2 rows; 16 elems/lane, shfl_xor wave reduction
  for (int rr = 0; rr < 2; ++rr) {
    const int row = wave * 2 + rr;
    const float* rowp = ldsS[row];
    float vals[16];
    float m = -1e30f;
#pragma unroll
    for (int j = 0; j < 16; ++j) {
      vals[j] = rowp[lane * 16 + j];
      m = fmaxf(m, vals[j]);
    }
#pragma unroll
    for (int off = 16; off > 0; off >>= 1) m = fmaxf(m, __shfl_xor(m, off, 32));
    float s = 0.0f;
#pragma unroll
    for (int j = 0; j < 16; ++j) { vals[j] = __expf(vals[j] - m); s += vals[j]; }
#pragma unroll
    for (int off = 16; off > 0; off >>= 1) s += __shfl_xor(s, off, 32);
    const float inv = 1.0f / s;

    const int l = r0 + row;
    _Float16* pOut = probs16 + ((size_t)bh * L_ + l) * L_;
    float*    aOut = attn0_out + ((size_t)b * L_ + l) * L_;
#pragma unroll
    for (int j = 0; j < 16; ++j) {
      const int col = lane * 16 + j;
      const float p = vals[j] * inv;
      pOut[col] = (_Float16)p;
      if (h == 0) aOut[col] = p;
    }
  }
}

// ---------------------------------------------------------------------------
// LayerNorm (row = 1024), optional affine, f32 in -> f16 out
// ---------------------------------------------------------------------------
__global__ __launch_bounds__(256) void k_ln(const float* x, const float* g,
                                            const float* bb, _Float16* out) {
  __shared__ float sred[8];
  const size_t row = blockIdx.x;
  const float* xp = x + row * D_;
  const int t = threadIdx.x;

  float v[4];
  float s = 0.0f;
#pragma unroll
  for (int i = 0; i < 4; ++i) { v[i] = xp[t + 256 * i]; s += v[i]; }
#pragma unroll
  for (int off = 16; off > 0; off >>= 1) s += __shfl_xor(s, off, 32);
  if ((t & 31) == 0) sred[t >> 5] = s;
  __syncthreads();
  float tot = 0.0f;
#pragma unroll
  for (int i = 0; i < 8; ++i) tot += sred[i];
  const float mu = tot * (1.0f / D_);

  float s2 = 0.0f;
#pragma unroll
  for (int i = 0; i < 4; ++i) { const float d = v[i] - mu; s2 += d * d; }
#pragma unroll
  for (int off = 16; off > 0; off >>= 1) s2 += __shfl_xor(s2, off, 32);
  __syncthreads();
  if ((t & 31) == 0) sred[t >> 5] = s2;
  __syncthreads();
  float tot2 = 0.0f;
#pragma unroll
  for (int i = 0; i < 8; ++i) tot2 += sred[i];
  const float rs = rsqrtf(tot2 * (1.0f / D_) + 1e-5f);

  _Float16* op = out + row * D_;
#pragma unroll
  for (int i = 0; i < 4; ++i) {
    const int c = t + 256 * i;
    float xn = (v[i] - mu) * rs;
    if (g) xn = xn * g[c] + bb[c];
    op[c] = (_Float16)xn;
  }
}

__global__ void k_cast(const float* in, _Float16* out, int n) {
  for (int i = blockIdx.x * blockDim.x + threadIdx.x; i < n;
       i += gridDim.x * blockDim.x)
    out[i] = (_Float16)in[i];
}

__global__ void k_copy(const float* in, float* out, int n) {
  const int i = blockIdx.x * blockDim.x + threadIdx.x;
  if (i < n) out[i] = in[i];
}

// ---------------------------------------------------------------------------
extern "C" void kernel_launch(void* const* d_in, const int* in_sizes, int n_in,
                              void* d_out, int out_size, void* d_ws,
                              size_t ws_size, hipStream_t stream) {
  (void)in_sizes; (void)n_in; (void)out_size; (void)ws_size;

  const float* s     = (const float*)d_in[0];
  const float* z     = (const float*)d_in[1];
  const float* wq    = (const float*)d_in[2];
  const float* bq    = (const float*)d_in[3];
  const float* wk    = (const float*)d_in[4];
  const float* bk    = (const float*)d_in[5];
  const float* wv    = (const float*)d_in[6];
  const float* bv    = (const float*)d_in[7];
  const float* wo    = (const float*)d_in[8];
  const float* bo    = (const float*)d_in[9];
  const float* cbias = (const float*)d_in[10];
  const float* sharp = (const float*)d_in[11];
  const float* ln_g  = (const float*)d_in[12];
  const float* ln_b  = (const float*)d_in[13];
  const float* w1    = (const float*)d_in[14];
  const float* b1    = (const float*)d_in[15];
  const float* w2    = (const float*)d_in[16];
  const float* b2    = (const float*)d_in[17];

  char* ws = (char*)d_ws;
  const size_t MB = 1u << 20;
  // persistent f16 weights
  _Float16* wq16 = (_Float16*)(ws + 0 * MB);
  _Float16* wk16 = (_Float16*)(ws + 2 * MB);
  _Float16* wv16 = (_Float16*)(ws + 4 * MB);
  _Float16* wo16 = (_Float16*)(ws + 6 * MB);
  _Float16* w116 = (_Float16*)(ws + 8 * MB);
  _Float16* w216 = (_Float16*)(ws + 16 * MB);
  // activations (with reuse)
  _Float16* sln16   = (_Float16*)(ws + 24 * MB);   // 16MB
  _Float16* q16     = (_Float16*)(ws + 40 * MB);   // 16MB
  _Float16* k16     = (_Float16*)(ws + 56 * MB);   // 16MB
  _Float16* v16     = (_Float16*)(ws + 72 * MB);   // 16MB
  _Float16* probs16 = (_Float16*)(ws + 88 * MB);   // 32MB
  _Float16* ao16    = (_Float16*)(ws + 24 * MB);   // reuse sln16
  float*    s1      = (float*)   (ws + 40 * MB);   // 32MB, reuse q16/k16
  _Float16* h16     = (_Float16*)(ws + 72 * MB);   // reuse v16
  _Float16* m16     = (_Float16*)(ws + 88 * MB);   // 64MB, reuse probs16

  float* out_s    = (float*)d_out;                       // [B,L,D]
  float* out_z    = out_s + (size_t)M_ * D_;             // [B,256]
  float* out_attn = out_z + (size_t)B_ * 256;            // [B,L,L]

  // 1) weights -> f16
  k_cast<<<2048, 256, 0, stream>>>(wq, wq16, D_ * D_);
  k_cast<<<2048, 256, 0, stream>>>(wk, wk16, D_ * D_);
  k_cast<<<2048, 256, 0, stream>>>(wv, wv16, D_ * D_);
  k_cast<<<2048, 256, 0, stream>>>(wo, wo16, D_ * D_);
  k_cast<<<4096, 256, 0, stream>>>(w1, w116, D_ * DFF_);
  k_cast<<<4096, 256, 0, stream>>>(w2, w216, DFF_ * D_);

  // 2) parameter-free LN(s) -> f16
  k_ln<<<M_, 256, 0, stream>>>(s, nullptr, nullptr, sln16);

  // 3) Q,K,V projections (WMMA): [8192x1024]x[1024x1024]
  const dim3 gP(M_ / 128, D_ / 64);
  k_gemm<true, false, false, true, false><<<gP, 256, 0, stream>>>(
      sln16, D_, wq16, D_, bq, nullptr, 0, nullptr, 0, q16, D_, D_);
  k_gemm<true, false, false, true, false><<<gP, 256, 0, stream>>>(
      sln16, D_, wk16, D_, bk, nullptr, 0, nullptr, 0, k16, D_, D_);
  k_gemm<true, false, false, true, false><<<gP, 256, 0, stream>>>(
      sln16, D_, wv16, D_, bv, nullptr, 0, nullptr, 0, v16, D_, D_);

  // 4) fused scores + bias/sharpen + softmax
  k_attn<<<dim3(L_ / 16, B_ * H_), 256, 0, stream>>>(q16, k16, cbias, sharp,
                                                     probs16, out_attn);

  // 5) attn @ V  (batched over b,h)
  k_gemm_av<<<dim3(L_ / 128, HD_ / 64, B_ * H_), 256, 0, stream>>>(probs16, v16,
                                                                   ao16);

  // 6) output projection + residual -> s1 (f32)
  k_gemm<true, false, true, false, true><<<gP, 256, 0, stream>>>(
      ao16, D_, wo16, D_, bo, s, D_, s1, D_, nullptr, 0, D_);

  // 7) LN(s1; g,b) -> f16
  k_ln<<<M_, 256, 0, stream>>>(s1, ln_g, ln_b, h16);

  // 8) MLP up + ReLU: [8192x1024]x[1024x4096] -> f16
  k_gemm<true, true, false, true, false><<<dim3(M_ / 128, DFF_ / 64), 256, 0,
                                           stream>>>(
      h16, D_, w116, DFF_, b1, nullptr, 0, nullptr, 0, m16, DFF_, D_);

  // 9) MLP down + residual -> d_out s section (f32)
  k_gemm<true, false, true, false, true><<<gP, 256, 0, stream>>>(
      m16, DFF_, w216, D_, b2, s1, D_, out_s, D_, nullptr, 0, DFF_);

  // 10) z pass-through
  k_copy<<<(B_ * 256 + 255) / 256, 256, 0, stream>>>(z, out_z, B_ * 256);
}
